// GraphCast_86303072846449
// MI455X (gfx1250) — matrix operand order repaired and checked by reading, not compiled, not run on hardware
//
#include <hip/hip_runtime.h>

// ---------------------------------------------------------------------------
// GraphCast encode-process-decode GNN for gfx1250 (MI455X).
//  * every MLP matmul = bf16 WMMA (v_wmma_f32_16x16x32_bf16), f32 accumulate
//  * weights converted fp32->bf16 once per launch (L2-resident, half traffic)
//  * double-buffered LDS pipeline: B tiles via global_load_async_to_lds_b128
//    (ASYNCcnt domain), A tiles via b128 loads + native v_cvt bf16 pack +
//    ds_store_b128; one barrier per K-step, global traffic overlapped w/ WMMA
//  * gathers (src[send]/dst[recv]) fused into the GEMM A row indexing
//  * segment_sum via fp32 L2 atomics; LayerNorm+residual wave32 row kernel
// ---------------------------------------------------------------------------

typedef __attribute__((ext_vector_type(16))) __bf16 v16bf;
typedef __attribute__((ext_vector_type(8)))  __bf16 bf16x8;
typedef __attribute__((ext_vector_type(2)))  __bf16 bf16x2;
typedef __attribute__((ext_vector_type(8)))  float  v8f;

#define C_DIM 512
#define BM 128
#define BN 64
#define BK 32
#define ASTR 40   // bf16 elems; 80B row stride -> conflict-free ds_load_b128
#define BSTR 72   // bf16 elems; 144B row stride, 16B aligned

enum { FLAG_BETA = 1, FLAG_SILU = 2 };

// Native bf16 pack: lowers to v_cvt_pk_bf16_f32-class hardware conversion.
__device__ __forceinline__ unsigned pk2(float a, float b) {
  bf16x2 t;
  t.x = (__bf16)a;
  t.y = (__bf16)b;
  unsigned r;
  __builtin_memcpy(&r, &t, 4);
  return r;
}
__device__ __forceinline__ unsigned short bfbits(float f) {
  __bf16 h = (__bf16)f;
  unsigned short r;
  __builtin_memcpy(&r, &h, 2);
  return r;
}

union AFrag { v16bf v; bf16x8 h[2]; };

// fp32 -> bf16(u16) bulk convert, 8 elems/thread, b128 in / b128 out
__global__ __launch_bounds__(256) void gc_f32_to_bf16(
    const float* __restrict__ src, unsigned short* __restrict__ dst, long n)
{
  const long i = ((long)blockIdx.x * 256 + threadIdx.x) * 8;
  if (i + 8 <= n) {
    const float4 a = ((const float4*)(src + i))[0];
    const float4 b = ((const float4*)(src + i))[1];
    uint4 p;
    p.x = pk2(a.x, a.y); p.y = pk2(a.z, a.w);
    p.z = pk2(b.x, b.y); p.w = pk2(b.z, b.w);
    *(uint4*)(dst + i) = p;
  } else {
    for (long c = i; c < n; ++c) dst[c] = bfbits(src[c]);
  }
}

// Y[M,N] (ld=N) = epilogue( A[rowIdx][0:K] @ W[K,N] + bias )
// epilogue: optional += existing Y (BETA), optional SiLU.  W is bf16 (u16).
__global__ __launch_bounds__(256) void gc_gemm_bf16_wmma(
    const float* __restrict__ A, const int* __restrict__ rowIdx, int lda,
    const unsigned short* __restrict__ W, int ldw,
    const float* __restrict__ bias,
    float* __restrict__ Y, int M, int N, int K, int flags)
{
  __shared__ unsigned short Abuf[2][BM * ASTR];
  __shared__ unsigned short Bbuf[2][BK * BSTR];

  const int tid  = threadIdx.x;
  const int lane = tid & 31;
  const int wave = tid >> 5;
  const int wm   = (wave & 3) * 32;   // wave M offset (4 waves)
  const int wn   = (wave >> 2) * 32;  // wave N offset (2 waves)
  const int blockM = blockIdx.y * BM;
  const int blockN = blockIdx.x * BN;

  // A staging: 2 threads per row, 16 contiguous columns each
  const int arow  = tid >> 1;          // 0..127
  const int acol0 = (tid & 1) * 16;    // 0 or 16
  const int asrcRow = blockM + arow;
  const bool arowOK = asrcRow < M;
  size_t asrc = 0;
  if (arowOK) {
    const int r = rowIdx ? rowIdx[asrcRow] : asrcRow;   // fused gather
    asrc = (size_t)r * (size_t)lda;
  }
  unsigned* const aDst0 = (unsigned*)&Abuf[0][arow * ASTR + acol0];
  unsigned* const aDst1 = (unsigned*)&Abuf[1][arow * ASTR + acol0];

  // async B copy: wave w moves rows 4w..4w+3, lane -> (row, 16B col chunk)
  const int bR = (wave << 2) + (lane >> 3);             // 0..31
  const int bC = (lane & 7) * 8;                        // bf16 col
  const unsigned bLds0 = (unsigned)(size_t)(&Bbuf[0][bR * BSTR + bC]);
  const unsigned bLds1 = (unsigned)(size_t)(&Bbuf[1][bR * BSTR + bC]);

  // tail B staging: 8 threads/row, 8 cols each
  const int brow  = tid >> 3;
  const int bcol0 = (tid & 7) * 8;

  v8f acc[2][2] = {};
  const int l16 = lane & 15;
  const int hi  = lane >> 4;

  // Edge rows of A never change: zero both parities once, skip in the loop.
  if (!arowOK) {
    const uint4 z = {0u, 0u, 0u, 0u};
    ((uint4*)aDst0)[0] = z; ((uint4*)aDst0)[1] = z;
    ((uint4*)aDst1)[0] = z; ((uint4*)aDst1)[1] = z;
  }

  auto loadA = [&](int k0, float f[16]) {
    const float* ap = A + asrc + (size_t)(k0 + acol0);
#pragma unroll
    for (int c = 0; c < 16; ++c) f[c] = ap[c];   // clang merges -> b128
  };
  auto storeA = [&](const float f[16], int par) {
    uint4 p0, p1;
    p0.x = pk2(f[0], f[1]);   p0.y = pk2(f[2], f[3]);
    p0.z = pk2(f[4], f[5]);   p0.w = pk2(f[6], f[7]);
    p1.x = pk2(f[8], f[9]);   p1.y = pk2(f[10], f[11]);
    p1.z = pk2(f[12], f[13]); p1.w = pk2(f[14], f[15]);
    unsigned* d = par ? aDst1 : aDst0;
    ((uint4*)d)[0] = p0;      // 16B-aligned: 80*arow + 2*acol0
    ((uint4*)d)[1] = p1;
  };
  auto stageB = [&](int k0, int par) {   // CDNA5 async bf16 copy, b128/lane
    const unsigned short* gp = W + (size_t)(k0 + bR) * ldw + blockN + bC;
    const unsigned long long ga = (unsigned long long)(size_t)gp;
    const unsigned dst = par ? bLds1 : bLds0;
    asm volatile("global_load_async_to_lds_b128 %0, %1, off"
                 :: "v"(dst), "v"(ga) : "memory");
  };
  auto mma_tile = [&](int par) {
    const unsigned short* ab = Abuf[par];
    const unsigned short* bb = Bbuf[par];
    AFrag a[2], b[2];
#pragma unroll
    for (int i = 0; i < 2; ++i) {   // lanes 0-15: K [0,8)+[16,24); 16-31 rest
      const unsigned short* p = &ab[(wm + i * 16 + l16) * ASTR + hi * 8];
      a[i].h[0] = *(const bf16x8*)(p);
      a[i].h[1] = *(const bf16x8*)(p + 16);
    }
#pragma unroll
    for (int j = 0; j < 2; ++j) {   // lane = N col; K [0,8)+[16,24) for lanes 0-15, rest 16-31
      const unsigned short* p = &bb[(hi * 8) * BSTR + wn + j * 16 + l16];
#pragma unroll
      for (int s = 0; s < 8; ++s) {
        b[j].h[0][s] = __builtin_bit_cast(__bf16, p[s * BSTR]);
        b[j].h[1][s] = __builtin_bit_cast(__bf16, p[(16 + s) * BSTR]);
      }
    }
#pragma unroll
    for (int i = 0; i < 2; ++i)
#pragma unroll
      for (int j = 0; j < 2; ++j)
        acc[i][j] = __builtin_amdgcn_wmma_f32_16x16x32_bf16(
            false, a[i].v, false, b[j].v, (short)0, acc[i][j], false, false);
  };

  const int kFull = K & ~(BK - 1);
  const int nt = kFull >> 5;          // full K tiles

  // ---- pipelined main loop: 1 barrier / K-step, async B + reg-staged A ----
  if (nt > 0) {
    if (arowOK) { float f[16]; loadA(0, f); storeA(f, 0); }
    stageB(0, 0);
    asm volatile("s_wait_asynccnt 0x0" ::: "memory");
    __syncthreads();
  }
  for (int t = 0; t < nt; ++t) {
    const int cur = t & 1;
    const bool more = (t + 1 < nt);
    if (more) stageB((t + 1) << 5, cur ^ 1);
    float f[16];
    const bool doA = more && arowOK;
    if (doA) loadA((t + 1) << 5, f);
    if (arowOK && t + 2 < nt)
      __builtin_prefetch(A + asrc + (size_t)(((t + 2) << 5) + acol0), 0, 3);
    mma_tile(cur);                      // overlaps with the loads above
    if (doA) storeA(f, cur ^ 1);
    if (more) asm volatile("s_wait_asynccnt 0x0" ::: "memory");
    __syncthreads();
  }

  // ---- partial K tail (guarded scalar staging) ----
  if (kFull < K) {
    const int par = nt & 1;
    float f[16];
#pragma unroll
    for (int c = 0; c < 16; ++c) {
      const int kk = kFull + acol0 + c;
      f[c] = (arowOK && kk < K) ? A[asrc + kk] : 0.f;
    }
    storeA(f, par);
#pragma unroll
    for (int c = 0; c < 8; ++c) {
      const int kk = kFull + brow;
      const int nn = blockN + bcol0 + c;
      Bbuf[par][brow * BSTR + bcol0 + c] =
          (kk < K && nn < N) ? W[(size_t)kk * ldw + nn] : (unsigned short)0;
    }
    __syncthreads();
    mma_tile(par);
  }

  // Epilogue. C/D layout: VGPR r, lanes 0-15 -> M=r, lanes 16-31 -> M=r+8.
#pragma unroll
  for (int i = 0; i < 2; ++i) {
#pragma unroll
    for (int j = 0; j < 2; ++j) {
      const int n = blockN + wn + j * 16 + l16;
      if (n >= N) continue;
#pragma unroll
      for (int r = 0; r < 8; ++r) {
        const int m = blockM + wm + i * 16 + hi * 8 + r;
        if (m >= M) continue;
        float v = acc[i][j][r];
        if (bias) v += bias[n];
        const long off = (long)m * N + n;
        if (flags & FLAG_BETA) v += Y[off];
        if (flags & FLAG_SILU) v = v / (1.f + __expf(-v));
        Y[off] = v;
      }
    }
  }
}

// Row LayerNorm over C=512 with optional residual: Y = res + g*norm(X)+beta.
__global__ __launch_bounds__(256) void gc_layernorm_res(
    const float* __restrict__ X, const float* __restrict__ g,
    const float* __restrict__ beta, const float* __restrict__ res,
    float* __restrict__ Y, int M)
{
  const int lane = threadIdx.x & 31;
  const int row  = blockIdx.x * 8 + (threadIdx.x >> 5);
  if (row >= M) return;
  const float4* x4 = (const float4*)(X + (long)row * C_DIM);
  float4 v[4];
  float s = 0.f;
#pragma unroll
  for (int c = 0; c < 4; ++c) {
    v[c] = x4[c * 32 + lane];
    s += v[c].x + v[c].y + v[c].z + v[c].w;
  }
#pragma unroll
  for (int o = 16; o; o >>= 1) s += __shfl_xor(s, o, 32);
  const float mu = s * (1.f / 512.f);
  float q = 0.f;
#pragma unroll
  for (int c = 0; c < 4; ++c) {
    v[c].x -= mu; v[c].y -= mu; v[c].z -= mu; v[c].w -= mu;
    q += v[c].x * v[c].x + v[c].y * v[c].y + v[c].z * v[c].z + v[c].w * v[c].w;
  }
#pragma unroll
  for (int o = 16; o; o >>= 1) q += __shfl_xor(q, o, 32);
  const float rstd = rsqrtf(q * (1.f / 512.f) + 1e-5f);
  float4* y4 = (float4*)(Y + (long)row * C_DIM);
  const float4* r4 = res ? (const float4*)(res + (long)row * C_DIM) : nullptr;
#pragma unroll
  for (int c = 0; c < 4; ++c) {
    const int n4 = c * 32 + lane;
    const float4 gv = ((const float4*)g)[n4];
    const float4 bv = ((const float4*)beta)[n4];
    float4 o;
    o.x = gv.x * v[c].x * rstd + bv.x;
    o.y = gv.y * v[c].y * rstd + bv.y;
    o.z = gv.z * v[c].z * rstd + bv.z;
    o.w = gv.w * v[c].w * rstd + bv.w;
    if (r4) { const float4 rv = r4[n4]; o.x += rv.x; o.y += rv.y; o.z += rv.z; o.w += rv.w; }
    y4[n4] = o;
  }
}

// segment_sum: agg[recv[e]] += E[e]  (fp32 L2 atomics, 4 channels per thread)
__global__ __launch_bounds__(256) void gc_scatter_add(
    const float* __restrict__ E, const int* __restrict__ recv,
    float* __restrict__ agg, int Erows)
{
  const long tid = (long)blockIdx.x * blockDim.x + threadIdx.x;
  if (tid >= (long)Erows * 128) return;
  const int e  = (int)(tid >> 7);
  const int c4 = (int)(tid & 127) << 2;
  const int r  = recv[e];
  const float4 v = *(const float4*)(E + (long)e * C_DIM + c4);
  float* dst = agg + (long)r * C_DIM + c4;
  atomicAdd(dst + 0, v.x); atomicAdd(dst + 1, v.y);
  atomicAdd(dst + 2, v.z); atomicAdd(dst + 3, v.w);
}

// ---------------------------------------------------------------------------
// Host-side orchestration
// ---------------------------------------------------------------------------
namespace {

struct MlpF { const float *b1, *b2, *beta, *g, *w1, *w2; };                // raw fp32
struct MlpW { const float *b1, *b2, *beta, *g; const unsigned short *w1, *w2; };

inline void gemm(const float* A, const int* idx, int lda,
                 const unsigned short* W, int ldw, const float* bias, float* Y,
                 int M, int N, int K, int flags, hipStream_t s) {
  dim3 grid((N + BN - 1) / BN, (M + BM - 1) / BM);
  gc_gemm_bf16_wmma<<<grid, 256, 0, s>>>(A, idx, lda, W, ldw, bias, Y, M, N, K, flags);
}

inline void lnorm(const float* X, const float* g, const float* beta,
                  const float* res, float* Y, int M, hipStream_t s) {
  gc_layernorm_res<<<(M + 7) / 8, 256, 0, s>>>(X, g, beta, res, Y, M);
}

inline void run_mlp(const float* X, int M, int K, const MlpW& p,
                    const float* res, float* out, float* tH, float* tY,
                    hipStream_t s) {
  gemm(X,  nullptr, K,     p.w1, C_DIM, p.b1, tH, M, C_DIM, K,     FLAG_SILU, s);
  gemm(tH, nullptr, C_DIM, p.w2, C_DIM, p.b2, tY, M, C_DIM, C_DIM, 0,         s);
  lnorm(tY, p.g, p.beta, res, out, M, s);
}

inline void run_interaction(const MlpW& pe, const MlpW& pn,
                            const float* src, float* dst,
                            const int* send, const int* recv,
                            float* efeat, int E, int Ndst,
                            float* tH, float* tY, float* agg, hipStream_t s) {
  gemm(efeat, nullptr, C_DIM, pe.w1,                  C_DIM, pe.b1,   tH, E, C_DIM, C_DIM, 0,                    s);
  gemm(src,   send,    C_DIM, pe.w1 + 1L*C_DIM*C_DIM, C_DIM, nullptr, tH, E, C_DIM, C_DIM, FLAG_BETA,            s);
  gemm(dst,   recv,    C_DIM, pe.w1 + 2L*C_DIM*C_DIM, C_DIM, nullptr, tH, E, C_DIM, C_DIM, FLAG_BETA|FLAG_SILU,  s);
  gemm(tH,    nullptr, C_DIM, pe.w2,                  C_DIM, pe.b2,   tY, E, C_DIM, C_DIM, 0,                    s);
  lnorm(tY, pe.g, pe.beta, efeat, efeat, E, s);                 // e_new in place

  hipMemsetAsync(agg, 0, (size_t)Ndst * C_DIM * sizeof(float), s);
  {
    const long total = (long)E * 128;
    gc_scatter_add<<<(unsigned)((total + 255) / 256), 256, 0, s>>>(efeat, recv, agg, E);
  }

  gemm(dst, nullptr, C_DIM, pn.w1,                  C_DIM, pn.b1,   tH, Ndst, C_DIM, C_DIM, 0,                    s);
  gemm(agg, nullptr, C_DIM, pn.w1 + 1L*C_DIM*C_DIM, C_DIM, nullptr, tH, Ndst, C_DIM, C_DIM, FLAG_BETA|FLAG_SILU, s);
  gemm(tH,  nullptr, C_DIM, pn.w2,                  C_DIM, pn.b2,   tY, Ndst, C_DIM, C_DIM, 0,                    s);
  lnorm(tY, pn.g, pn.beta, dst, dst, Ndst, s);                  // n_new in place
}

inline MlpF mlp_at(void* const* d_in, int base) {
  MlpF p;
  p.b1   = (const float*)d_in[base + 0];
  p.b2   = (const float*)d_in[base + 1];
  p.beta = (const float*)d_in[base + 2];
  p.g    = (const float*)d_in[base + 3];
  p.w1   = (const float*)d_in[base + 4];
  p.w2   = (const float*)d_in[base + 5];
  return p;
}

} // namespace

extern "C" void kernel_launch(void* const* d_in, const int* in_sizes, int n_in,
                              void* d_out, int out_size, void* d_ws, size_t ws_size,
                              hipStream_t stream) {
  (void)in_sizes; (void)n_in; (void)out_size; (void)ws_size;

  const int N_GRID = 32768, N_MESH = 10242;
  const int E_G2M = 98304, E_MESH = 81920, E_M2G = 98304;
  const int IN_GRID = 474, IN_MESHN = 3, IN_EDGE = 4;

  // ---- inputs (setup_inputs order) ----
  const float* grid_nfeat = (const float*)d_in[0];
  const float* mesh_nfeat = (const float*)d_in[1];
  const float* g2m_efeat  = (const float*)d_in[2];
  const float* mesh_efeat = (const float*)d_in[3];
  const float* m2g_efeat  = (const float*)d_in[4];
  const int* g2m_send  = (const int*)d_in[5];
  const int* g2m_recv  = (const int*)d_in[6];
  const int* mesh_send = (const int*)d_in[7];
  const int* mesh_recv = (const int*)d_in[8];
  const int* m2g_send  = (const int*)d_in[9];
  const int* m2g_recv  = (const int*)d_in[10];

  // ---- params (jax pytree flatten: sorted dict keys) ----
  const MlpF f_dec_edge = mlp_at(d_in, 11);
  const MlpF f_dec_emb  = mlp_at(d_in, 17);
  const MlpF f_dec_node = mlp_at(d_in, 23);
  const MlpF f_emb_g2m  = mlp_at(d_in, 29);
  const MlpF f_emb_grid = mlp_at(d_in, 35);
  const MlpF f_emb_mesh = mlp_at(d_in, 41);
  const MlpF f_enc_edge = mlp_at(d_in, 47);
  const MlpF f_enc_grid = mlp_at(d_in, 53);
  const MlpF f_enc_node = mlp_at(d_in, 59);
  const float* out_b1   = (const float*)d_in[65];   // [512]
  const float* out_b2   = (const float*)d_in[66];   // [256]
  const float* out_w1f  = (const float*)d_in[67];   // [512,512]
  const float* out_w2f  = (const float*)d_in[68];   // [512,256]
  const MlpF f_proc_edge = mlp_at(d_in, 69);        // stacked [8,...]
  const MlpF f_proc_node = mlp_at(d_in, 75);        // stacked [8,...]
  const MlpF f_proc_emb  = mlp_at(d_in, 81);

  // ---- workspace carve-up ----
  float* ws = (float*)d_ws;
  size_t off = 0;
  float* grid   = ws + off; off += (size_t)N_GRID * C_DIM;
  float* mesh   = ws + off; off += (size_t)N_MESH * C_DIM;
  float* g2m_e  = ws + off; off += (size_t)E_G2M  * C_DIM;
  float* mesh_e = ws + off; off += (size_t)E_MESH * C_DIM;
  float* m2g_e  = ws + off; off += (size_t)E_M2G  * C_DIM;
  float* tH     = ws + off; off += (size_t)E_G2M  * C_DIM;
  float* tY     = ws + off; off += (size_t)E_G2M  * C_DIM;
  float* agg    = ws + off; off += (size_t)N_GRID * C_DIM;
  unsigned short* pool = (unsigned short*)(ws + off);   // bf16 weight pool

  auto cvt = [&](const float* src, size_t n) -> const unsigned short* {
    unsigned short* dst = pool;
    pool += (n + 7) & ~(size_t)7;                       // keep 16B alignment
    gc_f32_to_bf16<<<(unsigned)((n + 2047) / 2048), 256, 0, stream>>>(src, dst, (long)n);
    return dst;
  };
  auto cvt_mlp = [&](const MlpF& f, size_t n1, size_t n2) -> MlpW {
    MlpW p; p.b1 = f.b1; p.b2 = f.b2; p.beta = f.beta; p.g = f.g;
    p.w1 = cvt(f.w1, n1); p.w2 = cvt(f.w2, n2);
    return p;
  };

  const size_t CC = (size_t)C_DIM * C_DIM;
  const MlpW dec_edge = cvt_mlp(f_dec_edge, 3 * CC, CC);
  const MlpW dec_emb  = cvt_mlp(f_dec_emb,  (size_t)IN_EDGE * C_DIM, CC);
  const MlpW dec_node = cvt_mlp(f_dec_node, 2 * CC, CC);
  const MlpW emb_g2m  = cvt_mlp(f_emb_g2m,  (size_t)IN_EDGE * C_DIM, CC);
  const MlpW emb_grid = cvt_mlp(f_emb_grid, (size_t)IN_GRID * C_DIM, CC);
  const MlpW emb_mesh = cvt_mlp(f_emb_mesh, (size_t)IN_MESHN * C_DIM, CC);
  const MlpW enc_edge = cvt_mlp(f_enc_edge, 3 * CC, CC);
  const MlpW enc_grid = cvt_mlp(f_enc_grid, CC, CC);
  const MlpW enc_node = cvt_mlp(f_enc_node, 2 * CC, CC);
  const unsigned short* out_w1 = cvt(out_w1f, CC);
  const unsigned short* out_w2 = cvt(out_w2f, (size_t)C_DIM * 256);
  const unsigned short* pe_w1 = cvt(f_proc_edge.w1, 8 * 3 * CC);
  const unsigned short* pe_w2 = cvt(f_proc_edge.w2, 8 * CC);
  const unsigned short* pn_w1 = cvt(f_proc_node.w1, 8 * 2 * CC);
  const unsigned short* pn_w2 = cvt(f_proc_node.w2, 8 * CC);
  const MlpW proc_emb = cvt_mlp(f_proc_emb, (size_t)IN_EDGE * C_DIM, CC);

  // ======================= Encoder =======================
  run_mlp(grid_nfeat, N_GRID, IN_GRID,  emb_grid, nullptr, grid,  tH, tY, stream);
  run_mlp(mesh_nfeat, N_MESH, IN_MESHN, emb_mesh, nullptr, mesh,  tH, tY, stream);
  run_mlp(g2m_efeat,  E_G2M,  IN_EDGE,  emb_g2m,  nullptr, g2m_e, tH, tY, stream);
  run_interaction(enc_edge, enc_node, grid, mesh, g2m_send, g2m_recv,
                  g2m_e, E_G2M, N_MESH, tH, tY, agg, stream);
  run_mlp(grid, N_GRID, C_DIM, enc_grid, /*res=*/grid, grid, tH, tY, stream);

  // ======================= Processor =====================
  run_mlp(mesh_efeat, E_MESH, IN_EDGE, proc_emb, nullptr, mesh_e, tH, tY, stream);
  for (int b = 0; b < 8; ++b) {
    MlpW pe, pn;
    pe.b1 = f_proc_edge.b1 + (size_t)b * C_DIM;
    pe.b2 = f_proc_edge.b2 + (size_t)b * C_DIM;
    pe.beta = f_proc_edge.beta + (size_t)b * C_DIM;
    pe.g  = f_proc_edge.g  + (size_t)b * C_DIM;
    pe.w1 = pe_w1 + (size_t)b * 3 * CC;
    pe.w2 = pe_w2 + (size_t)b * CC;
    pn.b1 = f_proc_node.b1 + (size_t)b * C_DIM;
    pn.b2 = f_proc_node.b2 + (size_t)b * C_DIM;
    pn.beta = f_proc_node.beta + (size_t)b * C_DIM;
    pn.g  = f_proc_node.g  + (size_t)b * C_DIM;
    pn.w1 = pn_w1 + (size_t)b * 2 * CC;
    pn.w2 = pn_w2 + (size_t)b * CC;
    run_interaction(pe, pn, mesh, mesh, mesh_send, mesh_recv,
                    mesh_e, E_MESH, N_MESH, tH, tY, agg, stream);
  }

  // ======================= Decoder =======================
  run_mlp(m2g_efeat, E_M2G, IN_EDGE, dec_emb, nullptr, m2g_e, tH, tY, stream);
  run_interaction(dec_edge, dec_node, mesh, grid, m2g_send, m2g_recv,
                  m2g_e, E_M2G, N_GRID, tH, tY, agg, stream);

  // Output head (no LayerNorm): d_out = silu(grid@w1+b1)@w2+b2  [32768,256]
  gemm(grid, nullptr, C_DIM, out_w1, C_DIM, out_b1, tH, N_GRID, C_DIM, C_DIM,
       FLAG_SILU, stream);
  gemm(tH, nullptr, C_DIM, out_w2, 256, out_b2, (float*)d_out, N_GRID, 256,
       C_DIM, 0, stream);
}
